// SocialLSTM_66322884985172
// MI455X (gfx1250) — matrix operand order repaired and checked by reading, not compiled, not run on hardware
//
#include <hip/hip_runtime.h>
#include <math.h>

// ---------------- problem constants ----------------
#define N_AG   2048
#define TT     8
#define HID    128
#define EMB    64
#define KH     512      // G*G*H  (social tensor flat dim)
#define KX     256      // concat GEMM K: [et(64)|at(64)|h(128)]
#define Z4     512      // 4*H gates
#define PRED_LEN 12
#define GRIDC  2
#define HALF_F 1.0f
#define EPS_F  0.01f
#define CELL_F 1.0f

#define AS1 __attribute__((address_space(1)))
#define AS3 __attribute__((address_space(3)))

#if __has_builtin(__builtin_amdgcn_global_load_async_to_lds_b128)
#define HAVE_ASYNC_LDS 1
#endif

// ---------------- types for WMMA ----------------
typedef __attribute__((ext_vector_type(16))) __bf16 v16bf;
typedef __attribute__((ext_vector_type(8)))  float  v8f;
typedef __attribute__((ext_vector_type(4)))  unsigned int uint4v;
typedef int v4i __attribute__((vector_size(16)));   // matches async-LDS builtin param

union FragAB { v16bf v; uint4v q[2]; };

// f32 <-> bf16 (RNE)
__device__ __forceinline__ unsigned short f2bf(float f) {
  union { float f; unsigned u; } v; v.f = f;
  unsigned r = v.u + 0x7FFFu + ((v.u >> 16) & 1u);
  return (unsigned short)(r >> 16);
}
__device__ __forceinline__ float bf2f(unsigned short b) {
  union { unsigned u; float f; } v; v.u = ((unsigned)b) << 16;
  return v.f;
}
__device__ __forceinline__ float sigm(float x) { return 1.0f / (1.0f + expf(-x)); }

// A fragment (16x32 bf16, MxK), row-major source with leading dim ld.
// ISA layout: lanes 0-15 hold row=lane, K = kc+{0..7, 16..23};
//             lanes 16-31 hold row=lane-16, K = kc+{8..15, 24..31}.
__device__ __forceinline__ v16bf load_frag_a(const unsigned short* base, int ld,
                                             int row0, int kc, int lane) {
  int m  = row0 + (lane & 15);
  int kh = (lane & 16) ? 8 : 0;
  const unsigned short* p = base + m * ld + kc;
  FragAB f;
  f.q[0] = *(const uint4v*)(p + kh);
  f.q[1] = *(const uint4v*)(p + 16 + kh);
  return f.v;
}

// B fragment (32x16 bf16, KxN) taken from a row-major weight W[n][k]
// (i.e. computing X @ W^T). lanes 0-15: col=lane, K=kc+{0..7, 16..23};
// lanes 16-31: col=lane-16, K=kc+{8..15, 24..31}.
__device__ __forceinline__ v16bf load_frag_b(const unsigned short* W, int ld,
                                             int col0, int kc, int lane) {
  int n  = col0 + (lane & 15);
  int kh = (lane & 16) ? 8 : 0;
  const unsigned short* p = W + n * ld + kc + kh;
  FragAB f;
  f.q[0] = *(const uint4v*)(p);
  f.q[1] = *(const uint4v*)(p + 16);
  return f.v;
}

// ---------------- setup kernels ----------------
__global__ void init_kernel(const float* __restrict__ h0, const float* __restrict__ c0,
                            const float* __restrict__ traj,
                            float* __restrict__ hf, float* __restrict__ cf,
                            unsigned short* __restrict__ hbf, float* __restrict__ point) {
  int idx = blockIdx.x * blockDim.x + threadIdx.x;
  if (idx < N_AG * HID) {
    int k = idx & (HID - 1);
    float h = h0[k], c = c0[k];
    hf[idx] = h; cf[idx] = c; hbf[idx] = f2bf(h);
  }
  if (idx < N_AG * 2) {
    int i = idx >> 1, j = idx & 1;
    point[idx] = traj[(i * TT + (TT - 1)) * 2 + j];
  }
}

__global__ void pack_kernel(const float* __restrict__ W_ih, const float* __restrict__ W_hh,
                            const float* __restrict__ soc_W,
                            const float* __restrict__ b_ih, const float* __restrict__ b_hh,
                            unsigned short* __restrict__ Wzb, unsigned short* __restrict__ socWb,
                            float* __restrict__ zb) {
  int idx = blockIdx.x * blockDim.x + threadIdx.x;
  if (idx < Z4 * KX) {
    int n = idx / KX, k = idx % KX;
    float v = (k < HID) ? W_ih[n * HID + k] : W_hh[n * HID + (k - HID)];
    Wzb[idx] = f2bf(v);
  } else if (idx < Z4 * KX + EMB * KH) {
    int j = idx - Z4 * KX;
    socWb[j] = f2bf(soc_W[j]);
  } else if (idx < Z4 * KX + EMB * KH + Z4) {
    int n = idx - Z4 * KX - EMB * KH;
    zb[n] = b_ih[n] + b_hh[n];
  }
}

// scene IDs are sorted -> contiguous segments; local scan (~32 iters).
__global__ void seg_kernel(const int* __restrict__ scene, int* __restrict__ seg) {
  int i = blockIdx.x * blockDim.x + threadIdx.x;
  if (i >= N_AG) return;
  int s = scene[i];
  int lo = i;
  while (lo > 0 && scene[lo - 1] == s) --lo;
  int hi = i + 1;
  while (hi < N_AG && scene[hi] == s) ++hi;
  seg[2 * i] = lo; seg[2 * i + 1] = hi;
}

// ---------------- social pooling ----------------
// one block per agent, thread = hidden index; writes Ht as bf16 [N][512]
__global__ void __launch_bounds__(128)
pool_kernel(const unsigned short* __restrict__ hbf,
            const float* __restrict__ pos, int pos_ld,
            const int* __restrict__ amask, int amask_ld,
            const int* __restrict__ seg,
            unsigned short* __restrict__ Htb) {
  const int i = blockIdx.x;
  const int t = threadIdx.x;   // hidden index 0..127
  float a0 = 0.f, a1 = 0.f, a2 = 0.f, a3 = 0.f;
  if (amask[i * amask_ld] == 1) {
    const float px = pos[i * pos_ld], py = pos[i * pos_ld + 1];
    const int s = seg[2 * i], e = seg[2 * i + 1];
    for (int j = s; j < e; ++j) {
      if (j == i) continue;
      if (amask[j * amask_ld] != 1) continue;
      float rx = pos[j * pos_ld]     - px;
      float ry = pos[j * pos_ld + 1] - py;
      if (rx < HALF_F - EPS_F && rx > -(HALF_F - EPS_F) &&
          ry < HALF_F - EPS_F && ry > -(HALF_F - EPS_F)) {
        int gx = (int)floorf((rx + HALF_F) / CELL_F);
        int gy = (int)floorf((ry + HALF_F) / CELL_F);
        int g = gx * GRIDC + gy;
        float hv = bf2f(hbf[j * HID + t]);
        if (g == 0) a0 += hv; else if (g == 1) a1 += hv;
        else if (g == 2) a2 += hv; else a3 += hv;
      }
    }
  }
  unsigned short* o = Htb + i * KH + t;
  o[0]       = f2bf(a0);
  o[HID]     = f2bf(a1);
  o[2 * HID] = f2bf(a2);
  o[3 * HID] = f2bf(a3);
}

// ---------------- fused embed + at-GEMM + LSTM-gate GEMM + cell ----------------
// block = 256 threads (8 waves) handles 16 agent rows.
// Stage 0: async-copy the Ht tile (16x512 bf16, 16KB) into LDS (ASYNCcnt path);
//          4 waves would otherwise re-read it 4x from L2.
// Stage A: waves 0-3 compute at[16,64] = relu(Ht @ socW^T + soc_b) (WMMA, K=512, A from LDS)
// Stage B: all threads compute et[16,64] (K=2, VALU) and copy h rows to LDS
// Stage C: wave w owns gate columns 16w..16w+15 of i/f/g/o (tiles w, w+8, w+16, w+24)
//          z = [et|at|h](16x256) @ Wz^T -> 4 accumulators/wave, elementwise LSTM in-register.
__global__ void __launch_bounds__(256)
lstm_step_kernel(const unsigned short* __restrict__ Htb,
                 const unsigned short* __restrict__ socWb,
                 const float* __restrict__ soc_b,
                 const float* __restrict__ rel, int rel_ld,
                 const float* __restrict__ posW, const float* __restrict__ posB,
                 const unsigned short* __restrict__ Wzb,
                 const float* __restrict__ zbias,
                 unsigned short* __restrict__ hbf,
                 float* __restrict__ hf, float* __restrict__ cf,
                 const int* __restrict__ amask, int amask_ld) {
  __shared__ __align__(16) unsigned short xht[16][KH];  // staged Ht tile (16KB)
  __shared__ __align__(16) unsigned short xa[16][KX];   // [et|at|h] bf16 (8KB)
  __shared__ int am[16];

  const int tid  = threadIdx.x;
  const int lane = tid & 31;
  const int wv   = tid >> 5;          // 0..7
  const int row0 = blockIdx.x * 16;

  if (tid < 16) am[tid] = amask ? amask[(row0 + tid) * amask_ld] : 1;

  // ---- stage 0: stage Ht tile into LDS via async load-to-LDS ----
  {
    const unsigned short* gbase = Htb + row0 * KH;
    unsigned short* lbase = &xht[0][0];
    #pragma unroll
    for (int s = 0; s < 4; ++s) {
      int ch = (tid + s * 256) * 8;   // ushort index of 16B chunk (1024 chunks)
#ifdef HAVE_ASYNC_LDS
      __builtin_amdgcn_global_load_async_to_lds_b128(
          (AS1 v4i*)(size_t)(gbase + ch),
          (AS3 v4i*)(lbase + ch), 0, 0);
#else
      *(uint4v*)(lbase + ch) = *(const uint4v*)(gbase + ch);
#endif
    }
  }

  // ---- stage B: position embedding et -> xa[:,0:64] (overlaps async copy) ----
  {
    int e  = tid & 63;
    int mb = tid >> 6;                // 0..3
    float w0 = posW[e * 2 + 0], w1 = posW[e * 2 + 1], b = posB[e];
    #pragma unroll
    for (int s = 0; s < 4; ++s) {
      int m = mb + s * 4;
      float x0 = rel[(row0 + m) * rel_ld + 0];
      float x1 = rel[(row0 + m) * rel_ld + 1];
      float v = fmaf(x0, w0, fmaf(x1, w1, b));
      xa[m][e] = f2bf(v > 0.f ? v : 0.f);
    }
  }
  // ---- copy current h rows (bf16) -> xa[:,128:256] ----
  {
    #pragma unroll
    for (int s = 0; s < 8; ++s) {
      int idx = tid + s * 256;        // 16*128 elems
      int m = idx >> 7, k = idx & 127;
      xa[m][128 + k] = hbf[(row0 + m) * HID + k];
    }
  }

#ifdef HAVE_ASYNC_LDS
#if __has_builtin(__builtin_amdgcn_s_wait_asynccnt)
  __builtin_amdgcn_s_wait_asynccnt(0);
#else
  asm volatile("s_wait_asynccnt 0" ::: "memory");
#endif
#endif
  __syncthreads();

  // ---- stage A: social embedding at -> xa[:,64:128] (waves 0..3, WMMA K=512) ----
  if (wv < 4) {
    v8f acc = {};
    const unsigned short* xhtp = &xht[0][0];
    #pragma unroll
    for (int kc = 0; kc < KH; kc += 32) {
      v16bf a  = load_frag_a(xhtp, KH, 0, kc, lane);
      v16bf bb = load_frag_b(socWb, KH, wv * 16, kc, lane);
      acc = __builtin_amdgcn_wmma_f32_16x16x32_bf16(false, a, false, bb,
                                                    (short)0, acc, false, false);
    }
    int n = wv * 16 + (lane & 15);
    float bias = soc_b[n];
    int mh = (lane & 16) ? 8 : 0;
    #pragma unroll
    for (int r = 0; r < 8; ++r) {
      float v = acc[r] + bias;
      xa[r + mh][64 + n] = f2bf(v > 0.f ? v : 0.f);
    }
  }
  __syncthreads();

  // ---- stage C: gate GEMM, K=256, 4 gate tiles per wave ----
  v8f ci = {}, cfr = {}, cg = {}, co = {};
  const unsigned short* xap = &xa[0][0];
  #pragma unroll
  for (int kc = 0; kc < KX; kc += 32) {
    v16bf a  = load_frag_a(xap, KX, 0, kc, lane);
    v16bf b0 = load_frag_b(Wzb, KX, 0 * HID + wv * 16, kc, lane);
    ci  = __builtin_amdgcn_wmma_f32_16x16x32_bf16(false, a, false, b0, (short)0, ci,  false, false);
    v16bf b1 = load_frag_b(Wzb, KX, 1 * HID + wv * 16, kc, lane);
    cfr = __builtin_amdgcn_wmma_f32_16x16x32_bf16(false, a, false, b1, (short)0, cfr, false, false);
    v16bf b2 = load_frag_b(Wzb, KX, 2 * HID + wv * 16, kc, lane);
    cg  = __builtin_amdgcn_wmma_f32_16x16x32_bf16(false, a, false, b2, (short)0, cg,  false, false);
    v16bf b3 = load_frag_b(Wzb, KX, 3 * HID + wv * 16, kc, lane);
    co  = __builtin_amdgcn_wmma_f32_16x16x32_bf16(false, a, false, b3, (short)0, co,  false, false);
  }

  // ---- elementwise LSTM cell + masked state update ----
  const int n  = wv * 16 + (lane & 15);
  const float bi = zbias[0 * HID + n];
  const float bf = zbias[1 * HID + n];
  const float bg = zbias[2 * HID + n];
  const float bo = zbias[3 * HID + n];
  const int mh = (lane & 16) ? 8 : 0;
  #pragma unroll
  for (int r = 0; r < 8; ++r) {
    int m = r + mh;
    if (am[m]) {
      int rowg = row0 + m;
      float zi = ci[r]  + bi;
      float zf = cfr[r] + bf;
      float zg = cg[r]  + bg;
      float zo = co[r]  + bo;
      float cold = cf[rowg * HID + n];
      float cnew = sigm(zf) * cold + sigm(zi) * tanhf(zg);
      float hnew = sigm(zo) * tanhf(cnew);
      cf[rowg * HID + n]  = cnew;
      hf[rowg * HID + n]  = hnew;
      hbf[rowg * HID + n] = f2bf(hnew);
    }
  }
}

// ---------------- prediction head ----------------
__global__ void __launch_bounds__(256)
predout_kernel(const float* __restrict__ hf,
               const float* __restrict__ predW, const float* __restrict__ predB,
               float* __restrict__ point, float* __restrict__ rel2,
               float* __restrict__ outp,
               const int* __restrict__ tmask, const int* __restrict__ predm, int sstep) {
  int i = blockIdx.x * blockDim.x + threadIdx.x;
  if (i >= N_AG) return;
  float acc[5];
  #pragma unroll
  for (int c = 0; c < 5; ++c) acc[c] = predB[c];
  for (int k = 0; k < HID; ++k) {
    float hv = hf[i * HID + k];
    #pragma unroll
    for (int c = 0; c < 5; ++c) acc[c] = fmaf(hv, predW[c * HID + k], acc[c]);
  }
  point[i * 2]     += acc[0];
  point[i * 2 + 1] += acc[1];
  rel2[i * 2]     = acc[0];
  rel2[i * 2 + 1] = acc[1];
  float valid = (tmask[i * TT + TT - 1] == 1 && predm[i] == 1) ? 1.f : 0.f;
  #pragma unroll
  for (int c = 0; c < 5; ++c)
    outp[(i * PRED_LEN + sstep) * 5 + c] = acc[c] * valid;
}

// ---------------- launch ----------------
extern "C" void kernel_launch(void* const* d_in, const int* in_sizes, int n_in,
                              void* d_out, int out_size, void* d_ws, size_t ws_size,
                              hipStream_t stream) {
  const float* traj     = (const float*)d_in[0];
  const float* traj_rel = (const float*)d_in[1];
  const float* h0       = (const float*)d_in[2];
  const float* c0       = (const float*)d_in[3];
  const float* pos_W    = (const float*)d_in[4];
  const float* pos_b    = (const float*)d_in[5];
  const float* soc_W    = (const float*)d_in[6];
  const float* soc_b    = (const float*)d_in[7];
  const float* W_ih     = (const float*)d_in[8];
  const float* W_hh     = (const float*)d_in[9];
  const float* b_ih     = (const float*)d_in[10];
  const float* b_hh     = (const float*)d_in[11];
  const float* pred_W   = (const float*)d_in[12];
  const float* pred_b   = (const float*)d_in[13];
  const int*   tmask    = (const int*)d_in[14];
  const int*   predm    = (const int*)d_in[15];
  const int*   scene    = (const int*)d_in[16];
  float* outp = (float*)d_out;

  char* ws = (char*)d_ws;
  size_t off = 0;
  auto alloc = [&](size_t bytes) {
    void* p = ws + off;
    off = (off + bytes + 255) & ~(size_t)255;
    return p;
  };
  float*          hf    = (float*)         alloc((size_t)N_AG * HID * 4);
  float*          cfbuf = (float*)         alloc((size_t)N_AG * HID * 4);
  unsigned short* hbf   = (unsigned short*)alloc((size_t)N_AG * HID * 2);
  unsigned short* Htb   = (unsigned short*)alloc((size_t)N_AG * KH * 2);
  unsigned short* socWb = (unsigned short*)alloc((size_t)EMB * KH * 2);
  unsigned short* Wzb   = (unsigned short*)alloc((size_t)Z4 * KX * 2);
  float*          zb    = (float*)         alloc((size_t)Z4 * 4);
  int*            seg   = (int*)           alloc((size_t)N_AG * 2 * 4);
  float*          point = (float*)         alloc((size_t)N_AG * 2 * 4);
  float*          rel2  = (float*)         alloc((size_t)N_AG * 2 * 4);

  // setup (re-run every call: deterministic, no leftover state)
  init_kernel<<<(N_AG * HID + 255) / 256, 256, 0, stream>>>(h0, c0, traj, hf, cfbuf, hbf, point);
  {
    int total = Z4 * KX + EMB * KH + Z4;
    pack_kernel<<<(total + 255) / 256, 256, 0, stream>>>(W_ih, W_hh, soc_W, b_ih, b_hh,
                                                         Wzb, socWb, zb);
  }
  seg_kernel<<<(N_AG + 255) / 256, 256, 0, stream>>>(scene, seg);

  // observation phase
  for (int t = 0; t < TT; ++t) {
    pool_kernel<<<N_AG, 128, 0, stream>>>(hbf, traj + t * 2, TT * 2,
                                          tmask + t, TT, seg, Htb);
    lstm_step_kernel<<<N_AG / 16, 256, 0, stream>>>(Htb, socWb, soc_b,
                                                    traj_rel + t * 2, TT * 2,
                                                    pos_W, pos_b, Wzb, zb,
                                                    hbf, hf, cfbuf,
                                                    tmask + t, TT);
  }
  // prediction phase
  for (int s = 0; s < PRED_LEN; ++s) {
    predout_kernel<<<N_AG / 256, 256, 0, stream>>>(hf, pred_W, pred_b, point, rel2,
                                                   outp, tmask, predm, s);
    pool_kernel<<<N_AG, 128, 0, stream>>>(hbf, point, 2,
                                          tmask + (TT - 1), TT, seg, Htb);
    lstm_step_kernel<<<N_AG / 16, 256, 0, stream>>>(Htb, socWb, soc_b,
                                                    rel2, 2,
                                                    pos_W, pos_b, Wzb, zb,
                                                    hbf, hf, cfbuf,
                                                    nullptr, 0);
  }
}
